// FastHelgasonLayer_19567871001169
// MI455X (gfx1250) — matrix operand order repaired and checked by reading, not compiled, not run on hardware
//
#include <hip/hip_runtime.h>
#include <hip/hip_bf16.h>
#include <math.h>

// ---------------------------------------------------------------------------
// FastPoincareFourier MLP, rank-32 factorized, bf16 WMMA on gfx1250 (wave32)
//   W = F(rows x 32) @ G(32 x cols)  exactly (16-sparse spectrum => rank-32)
//   x->U1->h->U2->out with split-K for the skinny reductions.
// ---------------------------------------------------------------------------

typedef __attribute__((ext_vector_type(16))) __bf16 v16bf;
typedef __attribute__((ext_vector_type(8)))  float  v8f;

#if defined(__AMDGCN__) && __has_builtin(__builtin_amdgcn_global_load_async_to_lds_b128)
#define HAVE_ASYNC_LDS 1
typedef int v4i_t __attribute__((vector_size(16)));            // matches builtin pointee
typedef __attribute__((address_space(1))) v4i_t* as1_v4i;      // global ("__device__")
typedef __attribute__((address_space(3))) v4i_t* as3_v4i;      // LDS
#else
#define HAVE_ASYNC_LDS 0
#endif

#define D_MODEL 2048
#define D_FF    8192
#define TOTAL_W (D_MODEL * D_FF)
#define LHALF   (TOTAL_W / 2)          // 8388608 == 2^23
#define NC      16
#define ROWS    (2 * 4096)             // batch * seq

__device__ __forceinline__ unsigned short f2bf(float f) {
    unsigned u = __float_as_uint(f);
    unsigned r = 0x7FFFu + ((u >> 16) & 1u);   // round-to-nearest-even
    return (unsigned short)((u + r) >> 16);
}

// --------------------------- x: fp32 -> bf16 --------------------------------
__global__ void k_f32_to_bf16(const float* __restrict__ in,
                              unsigned short* __restrict__ out, int n) {
    int i = blockIdx.x * blockDim.x + threadIdx.x;
    if (i < n) out[i] = f2bf(in[i]);
}

// ------------------- F factor: rows x 32, carries coeff*scale/L -------------
__global__ void k_gen_F(const float* __restrict__ cr, const float* __restrict__ ci,
                        const int* __restrict__ idx, const float* __restrict__ scale,
                        unsigned short* __restrict__ F, int rows, int halfcols) {
    int t = blockIdx.x * blockDim.x + threadIdx.x;   // t = r*16 + k
    int k = t & 15;
    int r = t >> 4;
    if (r >= rows) return;
    unsigned long long p = (unsigned long long)(unsigned)idx[k] *
                           ((unsigned long long)(unsigned)r * (unsigned long long)(unsigned)halfcols);
    unsigned m = (unsigned)(p & (unsigned long long)(LHALF - 1));
    float ang = (float)m * (6.28318530717958647692f / (float)LHALF);
    float s, c;
    __sincosf(ang, &s, &c);
    float fac = scale[0] * (1.0f / (float)LHALF);
    float a = cr[k] * fac, b = ci[k] * fac;
    F[r * 32 + 2 * k]     = f2bf(a * c - b * s);
    F[r * 32 + 2 * k + 1] = f2bf(a * s + b * c);
}

// ------------------- G factor: 32 x cols ------------------------------------
__global__ void k_gen_G(const int* __restrict__ idx,
                        unsigned short* __restrict__ G, int cols) {
    int t = blockIdx.x * blockDim.x + threadIdx.x;   // t = k*cols + c
    int c = t % cols;
    int k = t / cols;
    if (k >= NC) return;
    unsigned long long p = (unsigned long long)(unsigned)idx[k] * (unsigned long long)(unsigned)(c >> 1);
    unsigned m = (unsigned)(p & (unsigned long long)(LHALF - 1));
    float ang = (float)m * (6.28318530717958647692f / (float)LHALF);
    float s, co;
    __sincosf(ang, &s, &co);
    unsigned short g0, g1;
    if (c & 1) { g0 = f2bf(s);  g1 = f2bf(co); }
    else       { g0 = f2bf(co); g1 = f2bf(-s); }
    G[(2 * k)     * cols + c] = g0;
    G[(2 * k + 1) * cols + c] = g1;
}

// ----------------- split-K partial reduction (deterministic) ----------------
__global__ void k_reduce_splitk(const float* __restrict__ P,
                                unsigned short* __restrict__ out,
                                int n, int chunks) {
    int i = blockIdx.x * blockDim.x + threadIdx.x;
    if (i >= n) return;
    float s = 0.f;
    for (int z = 0; z < chunks; ++z) s += P[(size_t)z * n + i];
    out[i] = f2bf(s);
}

// ------------------------ tiled bf16 WMMA GEMM ------------------------------
// C[M,N] = A[M,K] @ B[K,N] (+bias, +gelu). gridDim.z = split-K chunks.
// MODE 0: no epilogue -> fp32 split-K partials (out + z*M*N)
// MODE 1: +bias, exact-erf GELU -> bf16
// MODE 2: +bias -> fp32
#define BM   128
#define BK   32
#define LDSP 40   // padded LDS row stride (bf16 elems): 80B rows, 16B aligned
#define EPS  36   // epilogue scratch row stride in dwords (144B, 16B aligned)

template<int BN, int MODE>
__global__ __launch_bounds__(256)
void k_gemm_wmma(const unsigned short* __restrict__ A,
                 const unsigned short* __restrict__ B,
                 const float* __restrict__ bias,
                 void* __restrict__ out,
                 int M, int N, int K) {
    constexpr int WAVES_N = BN / 32;        // 4 or 1
    constexpr int WAVES_M = 8 / WAVES_N;    // 2 or 8
    constexpr int MI = BM / (16 * WAVES_M); // 4 or 1
    constexpr int NI = 2;                   // 32 cols per wave

    constexpr int STAGE_BYTES = (BM * LDSP + BN * LDSP) * 2;
    constexpr int EP_BYTES    = 8 * 16 * EPS * 4;
    constexpr int SMEM_BYTES  = STAGE_BYTES > EP_BYTES ? STAGE_BYTES : EP_BYTES;
    __shared__ __align__(16) unsigned char smem[SMEM_BYTES];
    unsigned short* As = (unsigned short*)smem;
    unsigned short* Bs = As + BM * LDSP;    // transposed: [n][k]

    const int tid  = threadIdx.x;
    const int lane = tid & 31;
    const int w    = tid >> 5;
    const int wm   = w / WAVES_N;
    const int wn   = w % WAVES_N;
    const int half = lane >> 4;
    const int l16  = lane & 15;

    const long bm = (long)blockIdx.y * BM;
    const long bn = (long)blockIdx.x * BN;

    v8f acc[MI][NI];
    const v8f vzero = {0.f, 0.f, 0.f, 0.f, 0.f, 0.f, 0.f, 0.f};
    #pragma unroll
    for (int i = 0; i < MI; ++i)
        #pragma unroll
        for (int j = 0; j < NI; ++j)
            acc[i][j] = vzero;

    const int kper = (K / BK) / gridDim.z;
    const int k0   = blockIdx.z * kper;

    for (int kt = k0; kt < k0 + kper; ++kt) {
        // ---- stage A tile (BM x BK), no transpose -> async copy to LDS -----
#if HAVE_ASYNC_LDS
        #pragma unroll
        for (int v = 0; v < 2; ++v) {
            int vec = tid + v * 256;
            int row = vec >> 2;
            int kc  = (vec & 3) << 3;
            const unsigned short* gp = A + (bm + row) * (long)K + (long)kt * BK + kc;
            __builtin_amdgcn_global_load_async_to_lds_b128(
                (as1_v4i)(unsigned long long)(const void*)gp,
                (as3_v4i)(unsigned)(unsigned long long)(void*)&As[row * LDSP + kc],
                0, 0);
        }
#else
        #pragma unroll
        for (int v = 0; v < 2; ++v) {
            int vec = tid + v * 256;
            int row = vec >> 2;
            int kc  = (vec & 3) << 3;
            const uint4 d = *(const uint4*)(A + (bm + row) * (long)K + (long)kt * BK + kc);
            *(uint4*)(&As[row * LDSP + kc]) = d;
            if (kt + 1 < k0 + kper)
                __builtin_prefetch(A + (bm + row) * (long)K + (long)(kt + 1) * BK + kc, 0, 1);
        }
#endif
        // ---- stage B tile transposed (Bs[n][k]) : coalesced global reads ---
        constexpr int BVECS = (BK * BN) / 8;  // 512 or 128
        for (int vec = tid; vec < BVECS; vec += 256) {
            int krow = vec / (BN / 8);
            int nc   = (vec % (BN / 8)) * 8;
            union { uint4 u; unsigned short h[8]; } d;
            d.u = *(const uint4*)(B + (long)(kt * BK + krow) * N + bn + nc);
            #pragma unroll
            for (int e = 0; e < 8; ++e)
                Bs[(nc + e) * LDSP + krow] = d.h[e];
        }
#if HAVE_ASYNC_LDS
        asm volatile("s_wait_asynccnt 0x0" ::: "memory");
#endif
        __syncthreads();

        // ---- B fragments: 32x16 bf16, lane=N, VGPR j holds K = half*16 + 2j
        union { v16bf v; unsigned u[8]; } bfr[NI];
        #pragma unroll
        for (int ni = 0; ni < NI; ++ni) {
            int n = wn * 32 + ni * 16 + l16;
            #pragma unroll
            for (int j = 0; j < 8; ++j)
                bfr[ni].u[j] = *(const unsigned*)(&Bs[n * LDSP + half * 16 + j * 2]);
        }
        // ---- A fragments: 16x32 bf16: K = half*16 + j*2 ---
        #pragma unroll
        for (int mi = 0; mi < MI; ++mi) {
            union { v16bf v; unsigned u[8]; } af;
            int row = (wm * MI + mi) * 16 + l16;
            #pragma unroll
            for (int j = 0; j < 8; ++j) {
                int k = half * 16 + (j << 1);
                af.u[j] = *(const unsigned*)(&As[row * LDSP + k]);
            }
            #pragma unroll
            for (int ni = 0; ni < NI; ++ni)
                acc[mi][ni] = __builtin_amdgcn_wmma_f32_16x16x32_bf16(
                    false, af.v, false, bfr[ni].v, (short)0, acc[mi][ni], false, false);
        }
        __syncthreads();
    }

    // ---- vectorized epilogue via LDS bounce (smem reuse is barrier-safe) ---
    void* outp = out;
    if (MODE == 0)
        outp = (void*)((float*)out + (size_t)blockIdx.z * (size_t)M * (size_t)N);

    float* Ep = (float*)smem + (size_t)w * (16 * EPS);  // wave-private slice
    const int er  = lane >> 1;        // row within 16-row chunk
    const int ech = lane & 1;         // 16-col half

    #pragma unroll
    for (int mi = 0; mi < MI; ++mi) {
        #pragma unroll
        for (int ni = 0; ni < NI; ++ni) {
            float bv = 0.0f;
            if (MODE == 1 || MODE == 2) bv = bias[bn + wn * 32 + ni * 16 + l16];
            #pragma unroll
            for (int j = 0; j < 8; ++j) {
                float val = acc[mi][ni][j] + bv;
                if (MODE == 1)  // exact erf GELU
                    val = 0.5f * val * (1.0f + erff(val * 0.70710678118654752440f));
                Ep[(half * 8 + j) * EPS + ni * 16 + l16] = val;
            }
        }
        asm volatile("s_wait_dscnt 0x0" ::: "memory");  // wave-local RAW fence

        long grow = bm + (wm * MI + mi) * 16 + er;
        long gcol = bn + wn * 32 + ech * 16;
        const float* sf = Ep + er * EPS + ech * 16;     // 16 contiguous f32
        if (MODE == 1) {                                // bf16 out, 2x b128
            union { uint4 q[2]; unsigned short hh[16]; } o;
            #pragma unroll
            for (int e = 0; e < 16; ++e) o.hh[e] = f2bf(sf[e]);
            uint4* dst = (uint4*)((unsigned short*)outp + grow * (long)N + gcol);
            dst[0] = o.q[0];
            dst[1] = o.q[1];
        } else {                                        // fp32 out, 4x b128
            float4* dst = (float4*)((float*)outp + grow * (long)N + gcol);
            #pragma unroll
            for (int q = 0; q < 4; ++q) dst[q] = ((const float4*)sf)[q];
        }
    }
}

// ---------------------------------------------------------------------------
extern "C" void kernel_launch(void* const* d_in, const int* in_sizes, int n_in,
                              void* d_out, int out_size, void* d_ws, size_t ws_size,
                              hipStream_t stream) {
    const float* x        = (const float*)d_in[0];
    const float* fc_cr    = (const float*)d_in[1];
    const float* fc_ci    = (const float*)d_in[2];
    const float* fc_scale = (const float*)d_in[3];
    const float* pj_cr    = (const float*)d_in[4];
    const float* pj_ci    = (const float*)d_in[5];
    const float* pj_scale = (const float*)d_in[6];
    const float* fc_bias  = (const float*)d_in[7];
    const float* pj_bias  = (const float*)d_in[8];
    const int*   fc_idx   = (const int*)d_in[9];
    const int*   pj_idx   = (const int*)d_in[10];

    const int SK1 = 8;    // split-K for U1 (K=2048 -> 8 ksteps each)
    const int SK2 = 16;   // split-K for U2 (K=8192 -> 16 ksteps each)

    unsigned char* ws = (unsigned char*)d_ws;
    size_t off = 0;
    unsigned short* xb  = (unsigned short*)(ws + off); off += (size_t)ROWS * D_MODEL * 2;
    unsigned short* h   = (unsigned short*)(ws + off); off += (size_t)ROWS * D_FF * 2;
    unsigned short* Ffc = (unsigned short*)(ws + off); off += (size_t)D_MODEL * 32 * 2;
    unsigned short* Gfc = (unsigned short*)(ws + off); off += (size_t)32 * D_FF * 2;
    unsigned short* Fpj = (unsigned short*)(ws + off); off += (size_t)D_FF * 32 * 2;
    unsigned short* Gpj = (unsigned short*)(ws + off); off += (size_t)32 * D_MODEL * 2;
    unsigned short* U1  = (unsigned short*)(ws + off); off += (size_t)ROWS * 32 * 2;
    unsigned short* U2  = (unsigned short*)(ws + off); off += (size_t)ROWS * 32 * 2;
    float*          P1  = (float*)(ws + off);          off += (size_t)SK1 * ROWS * 32 * 4;
    float*          P2  = (float*)(ws + off);          off += (size_t)SK2 * ROWS * 32 * 4;

    // 1) x -> bf16
    {
        int n = ROWS * D_MODEL;
        k_f32_to_bf16<<<(n + 255) / 256, 256, 0, stream>>>(x, xb, n);
    }
    // 2) rank-32 factors of both weight matrices
    k_gen_F<<<(D_MODEL * 16 + 255) / 256, 256, 0, stream>>>(fc_cr, fc_ci, fc_idx, fc_scale,
                                                            Ffc, D_MODEL, D_FF / 2);
    k_gen_G<<<(NC * D_FF + 255) / 256, 256, 0, stream>>>(fc_idx, Gfc, D_FF);
    k_gen_F<<<(D_FF * 16 + 255) / 256, 256, 0, stream>>>(pj_cr, pj_ci, pj_idx, pj_scale,
                                                         Fpj, D_FF, D_MODEL / 2);
    k_gen_G<<<(NC * D_MODEL + 255) / 256, 256, 0, stream>>>(pj_idx, Gpj, D_MODEL);

    // 3) U1 = x @ F_fc   [8192 x 32], K=2048, split-K=8 -> fp32 partials
    {
        dim3 g(1, ROWS / BM, SK1);
        k_gemm_wmma<32, 0><<<g, 256, 0, stream>>>(xb, Ffc, nullptr, P1, ROWS, 32, D_MODEL);
        int n = ROWS * 32;
        k_reduce_splitk<<<(n + 255) / 256, 256, 0, stream>>>(P1, U1, n, SK1);
    }
    // 4) h = gelu(U1 @ G_fc + b_fc)   [8192 x 8192], K=32 (bandwidth-bound)
    {
        dim3 g(D_FF / 128, ROWS / BM, 1);
        k_gemm_wmma<128, 1><<<g, 256, 0, stream>>>(U1, Gfc, fc_bias, h, ROWS, D_FF, 32);
    }
    // 5) U2 = h @ F_pj   [8192 x 32], K=8192, split-K=16 -> fp32 partials
    {
        dim3 g(1, ROWS / BM, SK2);
        k_gemm_wmma<32, 0><<<g, 256, 0, stream>>>(h, Fpj, nullptr, P2, ROWS, 32, D_FF);
        int n = ROWS * 32;
        k_reduce_splitk<<<(n + 255) / 256, 256, 0, stream>>>(P2, U2, n, SK2);
    }
    // 6) out = U2 @ G_pj + b_proj   [8192 x 2048], K=32, fp32 out
    {
        dim3 g(D_MODEL / 128, ROWS / BM, 1);
        k_gemm_wmma<128, 2><<<g, 256, 0, stream>>>(U2, Gpj, pj_bias, d_out, ROWS, D_MODEL, 32);
    }
}
